// MultiDimensionalSpectralAttention_9045201125570
// MI455X (gfx1250) — matrix operand order repaired and checked by reading, not compiled, not run on hardware
//
#include <hip/hip_runtime.h>
#include <stdint.h>
#include <math.h>

// ---------------------------------------------------------------------------
// MultiDimensionalSpectralAttention for MI455X (gfx1250, wave32, WMMA)
// All contractions -> v_wmma_f32_16x16x32_bf16 (f32 accum).
// GEMM: double-buffered LDS, global_load_async_to_lds_b128 pipelined against
// WMMA (ASYNCcnt), global_prefetch two tiles ahead, one barrier per K-step.
// ---------------------------------------------------------------------------

#define Bq 2
#define Sq 2048
#define Eq 1024
#define Cq 16
#define Hq 16

typedef __attribute__((ext_vector_type(16))) __bf16 v16bf;
typedef __attribute__((ext_vector_type(8)))  float  v8f;

__device__ __forceinline__ uint16_t f2bf(float f) {
  uint32_t u = __float_as_uint(f);
  uint32_t r = (u + 0x7FFFu + ((u >> 16) & 1u)) >> 16;  // RNE
  return (uint16_t)r;
}

// ---------------------------------------------------------------------------
// Generic bf16 WMMA GEMM:  C = alpha * A(MxK) * op(B) + beta*C + bias
//   transB=1: B is NxK row-major (C = A*B^T)   -> async-copy staging
//   transB=0: B is KxN row-major               -> transposing ds staging
// Block tile 128x128x32, 256 threads = 8 waves, each wave 4x2 WMMA tiles.
// ---------------------------------------------------------------------------
#define BM 128
#define BN 128
#define BK 32
#define LDSP 20  // u32 per LDS row: 16 data (32 bf16) + 4 pad -> 80B, 16B aligned

union Frag { uint32_t u[8]; v16bf v; };

__global__ __launch_bounds__(256) void k_gemm_bf16(
    const uint16_t* __restrict__ A, const uint16_t* __restrict__ B,
    float* __restrict__ C, const float* __restrict__ bias,
    int M, int N, int K, int lda, int ldb, int ldc,
    long long sA, long long sB, long long sC,
    int transB, float alpha, float beta)
{
  __shared__ uint32_t As[2][BM][LDSP];
  __shared__ uint32_t Bs[2][BN][LDSP];

  int bz = blockIdx.z;
  A += (size_t)bz * sA;  B += (size_t)bz * sB;  C += (size_t)bz * sC;
  int bm = blockIdx.y * BM, bn = blockIdx.x * BN;
  int tid  = threadIdx.x;
  int lane = tid & 31, wave = tid >> 5;
  int wm = (wave & 1) * 64, wn = (wave >> 1) * 32;
  int lr = lane & 15, hi = lane >> 4;

  // per-thread staging coordinates (fully unrolled: 512 chunks / 256 threads)
  int r0 = tid >> 2,           c0 = tid & 3;          // chunk tid
  int r1 = (tid + 256) >> 2,   c1 = (tid + 256) & 3;  // chunk tid+256

  auto issueA = [&](int kt, int buf) {
    const uint16_t* g0 = A + (size_t)(bm + r0) * lda + kt + c0 * 8;
    uint32_t l0 = (uint32_t)(uintptr_t)&As[buf][r0][c0 * 4];
    asm volatile("global_load_async_to_lds_b128 %0, %1, off" :: "v"(l0), "v"(g0) : "memory");
    const uint16_t* g1 = A + (size_t)(bm + r1) * lda + kt + c1 * 8;
    uint32_t l1 = (uint32_t)(uintptr_t)&As[buf][r1][c1 * 4];
    asm volatile("global_load_async_to_lds_b128 %0, %1, off" :: "v"(l1), "v"(g1) : "memory");
  };
  auto issueB = [&](int kt, int buf) {
    const uint16_t* g0 = B + (size_t)(bn + r0) * ldb + kt + c0 * 8;
    uint32_t l0 = (uint32_t)(uintptr_t)&Bs[buf][r0][c0 * 4];
    asm volatile("global_load_async_to_lds_b128 %0, %1, off" :: "v"(l0), "v"(g0) : "memory");
    const uint16_t* g1 = B + (size_t)(bn + r1) * ldb + kt + c1 * 8;
    uint32_t l1 = (uint32_t)(uintptr_t)&Bs[buf][r1][c1 * 4];
    asm volatile("global_load_async_to_lds_b128 %0, %1, off" :: "v"(l1), "v"(g1) : "memory");
  };
  auto stageBt = [&](int kt, int buf) {  // B is KxN: transpose into Bs[n][k-pairs]
#pragma unroll
    for (int rep = 0; rep < 8; rep++) {
      int i = tid + rep * 256;           // BN*16 = 2048 u32 elements
      int n = i >> 4, kc = i & 15;
      uint32_t lo = B[(size_t)(kt + 2 * kc)     * ldb + bn + n];
      uint32_t hw = B[(size_t)(kt + 2 * kc + 1) * ldb + bn + n];
      Bs[buf][n][kc] = lo | (hw << 16);
    }
  };

  v8f acc[4][2];
#pragma unroll
  for (int i = 0; i < 4; i++)
#pragma unroll
    for (int j = 0; j < 2; j++)
#pragma unroll
      for (int r = 0; r < 8; r++) acc[i][j][r] = 0.0f;

  // prologue: stage tile 0 into buffer 0
  issueA(0, 0);
  if (transB) issueB(0, 0); else stageBt(0, 0);

  int buf = 0;
  for (int kt = 0; kt < K; kt += BK, buf ^= 1) {
    // publish tile kt (our async copies landed) and retire previous compute
    asm volatile("s_wait_asynccnt 0" ::: "memory");
    __syncthreads();

    // pipeline: issue tile kt+BK into the other buffer, prefetch kt+2*BK
    int nkt = kt + BK;
    if (nkt < K) {
      issueA(nkt, buf ^ 1);
      if (transB) issueB(nkt, buf ^ 1); else stageBt(nkt, buf ^ 1);
      int pkt = nkt + BK;
      if (pkt < K) {
        __builtin_prefetch(A + (size_t)(bm + (tid >> 1)) * lda + pkt + (tid & 1) * 16, 0, 3);
        if (transB)
          __builtin_prefetch(B + (size_t)(bn + (tid >> 1)) * ldb + pkt + (tid & 1) * 16, 0, 3);
      }
    }

    // ---- fragments (ISA 7.12.2 wave32 layouts) ----
    Frag aF[4], bF[2];
#pragma unroll
    for (int tm = 0; tm < 4; tm++) {
      int row = wm + tm * 16 + lr;
#pragma unroll
      for (int v = 0; v < 8; v++) {
        int kc = (v < 4) ? (v + hi * 4) : (v + 4 + hi * 4);
        aF[tm].u[v] = As[buf][row][kc];
      }
    }
#pragma unroll
    for (int tn = 0; tn < 2; tn++) {
      int row = wn + tn * 16 + lr;
#pragma unroll
      for (int v = 0; v < 8; v++) bF[tn].u[v] = Bs[buf][row][(v < 4) ? (v + hi * 4) : (v + 4 + hi * 4)];
    }
#pragma unroll
    for (int tm = 0; tm < 4; tm++)
#pragma unroll
      for (int tn = 0; tn < 2; tn++)
        acc[tm][tn] = __builtin_amdgcn_wmma_f32_16x16x32_bf16(
            false, aF[tm].v, false, bF[tn].v, (short)0, acc[tm][tn], false, false);
  }

  // ---- epilogue: C/D layout M = r + 8*hi, N = lane&15 ----
#pragma unroll
  for (int tm = 0; tm < 4; tm++)
#pragma unroll
    for (int tn = 0; tn < 2; tn++) {
      int col = bn + wn + tn * 16 + lr;
#pragma unroll
      for (int r = 0; r < 8; r++) {
        int row = bm + wm + tm * 16 + r + hi * 8;
        size_t idx = (size_t)row * ldc + col;
        float v = alpha * acc[tm][tn][r];
        if (beta != 0.0f) v += beta * C[idx];
        if (bias) v += bias[col];
        C[idx] = v;
      }
    }
}

// ---------------------------------------------------------------------------
__global__ void k_cvt_bf16(const float* __restrict__ s, uint16_t* __restrict__ d, size_t n) {
  size_t i = (size_t)blockIdx.x * 256 + threadIdx.x;
  if (i < n) d[i] = f2bf(s[i]);
}

// Radix-2 FFT/IFFT over S=2048 per (b,e) column, planar re/im.
__global__ __launch_bounds__(256) void k_fft2048(
    const float* __restrict__ srcRe, const float* __restrict__ srcIm,
    float* __restrict__ dstRe, float* __restrict__ dstIm,
    int inverse, float scale,
    long long srcS, long long srcB, long long dstS, long long dstB)
{
  __shared__ float re[2048];
  __shared__ float im[2048];
  int e = blockIdx.x % Eq, b = blockIdx.x / Eq;
  int tid = threadIdx.x;
  const float* pr = srcRe + (long long)b * srcB + e;
  const float* pi = srcIm ? srcIm + (long long)b * srcB + e : nullptr;
  for (int s = tid; s < 2048; s += 256) {
    int r = __brev((unsigned)s) >> 21;  // 11-bit reverse
    re[r] = pr[(long long)s * srcS];
    im[r] = pi ? pi[(long long)s * srcS] : 0.0f;
  }
  __syncthreads();
  float sgn = inverse ? 1.0f : -1.0f;
  for (int len = 2; len <= 2048; len <<= 1) {
    int half = len >> 1;
    for (int j = tid; j < 1024; j += 256) {
      int blk = j / half, pos = j - blk * half;
      int i0 = blk * len + pos, i1 = i0 + half;
      float ang = sgn * 6.283185307179586f * (float)pos / (float)len;
      float sn, cs; __sincosf(ang, &sn, &cs);
      float xr = re[i1], xi = im[i1];
      float tr = xr * cs - xi * sn, ti = xr * sn + xi * cs;
      float ur = re[i0], ui = im[i0];
      re[i0] = ur + tr; im[i0] = ui + ti;
      re[i1] = ur - tr; im[i1] = ui - ti;
    }
    __syncthreads();
  }
  float* qr = dstRe + (long long)b * dstB + e;
  float* qi = dstIm ? dstIm + (long long)b * dstB + e : nullptr;
  for (int s = tid; s < 2048; s += 256) {
    qr[(long long)s * dstS] = re[s] * scale;
    if (qi) qi[(long long)s * dstS] = im[s] * scale;
  }
}

// Full Haar pyramid over S=2048 per (b,e) column (in-place halving scheme).
__global__ __launch_bounds__(256) void k_haar2048(const float* __restrict__ src,
                                                  float* __restrict__ dst)
{
  __shared__ float buf[2048];
  int e = blockIdx.x % Eq, b = blockIdx.x / Eq;
  int tid = threadIdx.x;
  const float* p = src + (long long)b * Sq * Eq + e;
  for (int s = tid; s < 2048; s += 256) buf[s] = p[(long long)s * Eq];
  __syncthreads();
  const float is2 = 0.7071067811865476f;
  for (int L = 2048; L >= 2; L >>= 1) {
    int half = L >> 1;
    float av[4], dv[4];
    int c = 0;
    for (int i = tid; i < half; i += 256) {
      float ev = buf[2 * i], ov = buf[2 * i + 1];
      av[c] = (ev + ov) * is2; dv[c] = (ev - ov) * is2; c++;
    }
    __syncthreads();
    c = 0;
    for (int i = tid; i < half; i += 256) { buf[i] = av[c]; buf[half + i] = dv[c]; c++; }
    __syncthreads();
  }
  float* q = dst + (long long)b * Sq * Eq + e;
  for (int s = tid; s < 2048; s += 256) q[(long long)s * Eq] = buf[s];
}

// conv1d E->C, kernel 3, pad 1: one wave per (b,s,c), lanes reduce over e.
__global__ __launch_bounds__(256) void k_conv3(const float* __restrict__ x,
    const float* __restrict__ Wc, const float* __restrict__ bc, float* __restrict__ ch)
{
  int unit = blockIdx.x * 8 + (threadIdx.x >> 5);
  int lane = threadIdx.x & 31;
  int c = unit % Cq;
  int s = (unit / Cq) % Sq;
  int b = unit / (Cq * Sq);
  const float* xb = x + (long long)b * Sq * Eq;
  float sum = 0.f;
  for (int e = lane; e < Eq; e += 32) {
    const float* wp = Wc + ((long long)c * Eq + e) * 3;
#pragma unroll
    for (int k = 0; k < 3; k++) {
      int ss = s + k - 1;
      if (ss >= 0 && ss < Sq) sum += xb[(long long)ss * Eq + e] * wp[k];
    }
  }
#pragma unroll
  for (int o = 16; o > 0; o >>= 1) sum += __shfl_xor(sum, o, 32);
  if (lane == 0) ch[(long long)unit] = sum + bc[c];
}

__global__ void k_fw_softmax(const float* __restrict__ Fp, float* __restrict__ fw) {
  int e = blockIdx.x * 256 + threadIdx.x;
  if (e >= Eq) return;
  float m = -1e30f;
  for (int c = 0; c < Cq; c++) m = fmaxf(m, Fp[c * Eq + e]);
  float sum = 0.f;
  for (int c = 0; c < Cq; c++) sum += __expf(Fp[c * Eq + e] - m);
  float inv = 1.0f / sum;
  for (int c = 0; c < Cq; c++) fw[c * Eq + e] = __expf(Fp[c * Eq + e] - m) * inv;
}

// ch2[b,s,e] = sum_c ch[b,s,c]*fw[c,e]  (emit bf16: feeds WMMA only)
__global__ void k_mix(const float* __restrict__ ch, const float* __restrict__ fw,
                      uint16_t* __restrict__ out)
{
  long long i = (long long)blockIdx.x * 256 + threadIdx.x;
  int e = (int)(i % Eq);
  long long bs = i / Eq;
  float sum = 0.f;
#pragma unroll
  for (int c = 0; c < Cq; c++) sum += ch[bs * Cq + c] * fw[c * Eq + e];
  out[i] = f2bf(sum);
}

__global__ __launch_bounds__(256) void k_row_softmax(const float* __restrict__ src,
    uint16_t* __restrict__ dst, int N)
{
  __shared__ float red[256];
  long long row = blockIdx.x;
  const float* p = src + row * N;
  int tid = threadIdx.x;
  float m = -1e30f;
  for (int i = tid; i < N; i += 256) m = fmaxf(m, p[i]);
  red[tid] = m; __syncthreads();
  for (int s = 128; s > 0; s >>= 1) { if (tid < s) red[tid] = fmaxf(red[tid], red[tid + s]); __syncthreads(); }
  m = red[0]; __syncthreads();
  float sum = 0.f;
  for (int i = tid; i < N; i += 256) sum += __expf(p[i] - m);
  red[tid] = sum; __syncthreads();
  for (int s = 128; s > 0; s >>= 1) { if (tid < s) red[tid] += red[tid + s]; __syncthreads(); }
  float inv = 1.0f / red[0];
  uint16_t* q = dst + row * N;
  for (int i = tid; i < N; i += 256) q[i] = f2bf(__expf(p[i] - m) * inv);
}

// complex: w = softmax(Re), cw = w * scores/|scores|  (bf16 planar out)
__global__ __launch_bounds__(256) void k_cplx_softmax(const float* __restrict__ srcRe,
    const float* __restrict__ srcIm, uint16_t* __restrict__ dstRe,
    uint16_t* __restrict__ dstIm, int N)
{
  __shared__ float red[256];
  long long row = blockIdx.x;
  const float* pr = srcRe + row * N;
  const float* pi = srcIm + row * N;
  int tid = threadIdx.x;
  float m = -1e30f;
  for (int i = tid; i < N; i += 256) m = fmaxf(m, pr[i]);
  red[tid] = m; __syncthreads();
  for (int s = 128; s > 0; s >>= 1) { if (tid < s) red[tid] = fmaxf(red[tid], red[tid + s]); __syncthreads(); }
  m = red[0]; __syncthreads();
  float sum = 0.f;
  for (int i = tid; i < N; i += 256) sum += __expf(pr[i] - m);
  red[tid] = sum; __syncthreads();
  for (int s = 128; s > 0; s >>= 1) { if (tid < s) red[tid] += red[tid + s]; __syncthreads(); }
  float inv = 1.0f / red[0];
  for (int i = tid; i < N; i += 256) {
    float re = pr[i], im = pi[i];
    float w = __expf(re - m) * inv;
    float r = sqrtf(re * re + im * im);
    float cr = 1.f, ci = 0.f;
    if (r > 0.f) { cr = re / r; ci = im / r; }
    dstRe[row * N + i] = f2bf(w * cr);
    dstIm[row * N + i] = f2bf(w * ci);
  }
}

// 3-token MHA core: one thread per (b*s, head); L=3, hd=64.
__global__ void k_mha3(const float* __restrict__ qkv, uint16_t* __restrict__ o) {
  int i = blockIdx.x * 256 + threadIdx.x;   // < B*S*H
  int h = i % Hq;
  long long bs = i / Hq;
  const float* base = qkv + bs * 3 * 3072;
  const float* q0 = base + h * 64;
  const float* k0 = base + 1024 + h * 64;
  const float* v0 = base + 2048 + h * 64;
  float sc[3][3];
  for (int l = 0; l < 3; l++)
    for (int m = 0; m < 3; m++) {
      float s = 0.f;
      for (int d = 0; d < 64; d++) s += q0[l * 3072 + d] * k0[m * 3072 + d];
      sc[l][m] = s * 0.125f;
    }
  for (int l = 0; l < 3; l++) {
    float mx = fmaxf(sc[l][0], fmaxf(sc[l][1], sc[l][2]));
    float e0 = __expf(sc[l][0] - mx), e1 = __expf(sc[l][1] - mx), e2 = __expf(sc[l][2] - mx);
    float inv = 1.f / (e0 + e1 + e2);
    float w0 = e0 * inv, w1 = e1 * inv, w2 = e2 * inv;
    for (int d = 0; d < 64; d++) {
      float s = w0 * v0[d] + w1 * v0[3072 + d] + w2 * v0[2 * 3072 + d];
      o[(bs * 3 + l) * 1024 + h * 64 + d] = f2bf(s);
    }
  }
}

// DCT-II matrix D and its inverse I = D^T, generated directly in bf16.
__global__ void k_dct_mats(uint16_t* __restrict__ D, uint16_t* __restrict__ I) {
  long long idx = (long long)blockIdx.x * 256 + threadIdx.x;  // < S*S
  int n = (int)(idx % Sq), k = (int)(idx / Sq);
  const float PI = 3.14159265358979323846f;
  float v = cosf(PI * (float)k * (2.0f * (float)n + 1.0f) / (2.0f * (float)Sq));
  float sc = (k == 0) ? sqrtf(1.0f / (float)Sq) : sqrtf(2.0f / (float)Sq);
  uint16_t bv = f2bf(v * sc);
  D[(long long)k * Sq + n] = bv;
  I[(long long)n * Sq + k] = bv;
}

// ---------------------------------------------------------------------------
extern "C" void kernel_launch(void* const* d_in, const int* in_sizes, int n_in,
                              void* d_out, int out_size, void* d_ws, size_t ws_size,
                              hipStream_t stream)
{
  (void)in_sizes; (void)n_in; (void)out_size; (void)ws_size;
  const float* query = (const float*)d_in[0];
  const float* Wp[3] = {(const float*)d_in[1], (const float*)d_in[6],  (const float*)d_in[11]};
  const float* bp[3] = {(const float*)d_in[2], (const float*)d_in[7],  (const float*)d_in[12]};
  const float* Wc[3] = {(const float*)d_in[3], (const float*)d_in[8],  (const float*)d_in[13]};
  const float* bc[3] = {(const float*)d_in[4], (const float*)d_in[9],  (const float*)d_in[14]};
  const float* Fp[3] = {(const float*)d_in[5], (const float*)d_in[10], (const float*)d_in[15]};
  const float* mha_iw = (const float*)d_in[16];
  const float* mha_ib = (const float*)d_in[17];
  const float* mha_ow = (const float*)d_in[18];
  const float* mha_ob = (const float*)d_in[19];
  const float* fus_W  = (const float*)d_in[20];
  const float* fus_b  = (const float*)d_in[21];
  float* out = (float*)d_out;

  const long long MSE = (long long)Bq * Sq * Eq;  // 4,194,304
  const long long SS  = (long long)Sq * Sq;
  const long long SE  = (long long)Sq * Eq;

  char* wbase = (char*)d_ws;
  size_t off = 0;
  auto alloc = [&](size_t bytes) -> char* {
    char* p = wbase + off; off += (bytes + 255) & ~(size_t)255; return p;
  };
  // persistent
  uint16_t* qbf       = (uint16_t*)alloc((size_t)MSE * 2);
  uint16_t* wtbf      = (uint16_t*)alloc((size_t)3 * Eq * Eq * 2);
  float*    stacked   = (float*)   alloc((size_t)MSE * 3 * 4);
  uint16_t* stackedbf = (uint16_t*)alloc((size_t)MSE * 3 * 2);
  size_t mark = off;
  // domain scratch
  float*    xd   = (float*)   alloc((size_t)MSE * 4);
  uint16_t* xdbf = (uint16_t*)alloc((size_t)MSE * 2);
  float*    sre  = (float*)   alloc((size_t)MSE * 4);
  float*    simb = (float*)   alloc((size_t)MSE * 4);
  float*    chre = (float*)   alloc((size_t)Bq * Sq * Cq * 4);
  float*    chim = (float*)   alloc((size_t)Bq * Sq * Cq * 4);
  float*    fwb  = (float*)   alloc((size_t)Cq * Eq * 4);
  uint16_t* c2re = (uint16_t*)alloc((size_t)MSE * 2);
  uint16_t* c2im = (uint16_t*)alloc((size_t)MSE * 2);
  float*    scre = (float*)   alloc((size_t)Bq * SS * 4);
  float*    scim = (float*)   alloc((size_t)Bq * SS * 4);
  uint16_t* wre  = (uint16_t*)alloc((size_t)Bq * SS * 2);
  uint16_t* wim  = (uint16_t*)alloc((size_t)Bq * SS * 2);
  float*    atre = (float*)   alloc((size_t)MSE * 4);
  float*    atim = (float*)   alloc((size_t)MSE * 4);
  uint16_t* atbf = (uint16_t*)alloc((size_t)MSE * 2);
  uint16_t* Dbf  = (uint16_t*)alloc((size_t)SS * 2);
  uint16_t* Ibf  = (uint16_t*)alloc((size_t)SS * 2);
  // MHA scratch overlays (domain scratch is dead by then)
  off = mark;
  float*    qkvb   = (float*)   alloc((size_t)Bq * Sq * 3 * 3 * Eq * 4);  // 12288 x 3072
  uint16_t* obf    = (uint16_t*)alloc((size_t)MSE * 3 * 2);
  float*    corr   = (float*)   alloc((size_t)MSE * 3 * 4);
  uint16_t* corrbf = (uint16_t*)alloc((size_t)MSE * 3 * 2);

  auto cvt = [&](const float* s, uint16_t* d, long long n) {
    k_cvt_bf16<<<dim3((unsigned)((n + 255) / 256)), dim3(256), 0, stream>>>(s, d, (size_t)n);
  };
  auto gemm = [&](const uint16_t* A, const uint16_t* Bm, float* Cm, const float* bias,
                  int M, int N, int K, int lda, int ldb, int ldc,
                  long long sA, long long sB, long long sC, int batch,
                  int transB, float alpha, float beta) {
    k_gemm_bf16<<<dim3(N / 128, M / 128, batch), dim3(256), 0, stream>>>(
        A, Bm, Cm, bias, M, N, K, lda, ldb, ldc, sA, sB, sC, transB, alpha, beta);
  };

  cvt(query, qbf, MSE);
  const float iscale = 0.03125f;  // 1/sqrt(E)

  for (int d = 0; d < 3; d++) {
    // xd = query @ Wp^T + bp
    cvt(Wp[d], wtbf, (long long)Eq * Eq);
    gemm(qbf, wtbf, xd, bp[d], Bq * Sq, Eq, Eq, Eq, Eq, Eq, 0, 0, 0, 1, 1, 1.0f, 0.0f);
    k_fw_softmax<<<dim3(Eq / 256), dim3(256), 0, stream>>>(Fp[d], fwb);

    if (d == 0) {  // ---- Fourier ----
      k_fft2048<<<dim3(Bq * Eq), dim3(256), 0, stream>>>(xd, nullptr, sre, simb,
          0, 1.0f, Eq, SE, Eq, SE);
      k_conv3<<<dim3(Bq * Sq * Cq / 8), dim3(256), 0, stream>>>(sre,  Wc[d], bc[d], chre);
      k_conv3<<<dim3(Bq * Sq * Cq / 8), dim3(256), 0, stream>>>(simb, Wc[d], bc[d], chim);
      k_mix<<<dim3((unsigned)(MSE / 256)), dim3(256), 0, stream>>>(chre, fwb, c2re);
      k_mix<<<dim3((unsigned)(MSE / 256)), dim3(256), 0, stream>>>(chim, fwb, c2im);
      // complex scores (scaled): re = s(qr kr^T - qi ki^T), im = s(qr ki^T + qi kr^T)
      gemm(c2re, c2re, scre, nullptr, Sq, Sq, Eq, Eq, Eq, Sq, SE, SE, SS, Bq, 1,  iscale, 0.0f);
      gemm(c2im, c2im, scre, nullptr, Sq, Sq, Eq, Eq, Eq, Sq, SE, SE, SS, Bq, 1, -iscale, 1.0f);
      gemm(c2re, c2im, scim, nullptr, Sq, Sq, Eq, Eq, Eq, Sq, SE, SE, SS, Bq, 1,  iscale, 0.0f);
      gemm(c2im, c2re, scim, nullptr, Sq, Sq, Eq, Eq, Eq, Sq, SE, SE, SS, Bq, 1,  iscale, 1.0f);
      k_cplx_softmax<<<dim3(Bq * Sq), dim3(256), 0, stream>>>(scre, scim, wre, wim, Sq);
      // out = cw @ v (complex)
      gemm(wre, c2re, atre, nullptr, Sq, Eq, Sq, Sq, Eq, Eq, SS, SE, SE, Bq, 0,  1.0f, 0.0f);
      gemm(wim, c2im, atre, nullptr, Sq, Eq, Sq, Sq, Eq, Eq, SS, SE, SE, Bq, 0, -1.0f, 1.0f);
      gemm(wre, c2im, atim, nullptr, Sq, Eq, Sq, Sq, Eq, Eq, SS, SE, SE, Bq, 0,  1.0f, 0.0f);
      gemm(wim, c2re, atim, nullptr, Sq, Eq, Sq, Sq, Eq, Eq, SS, SE, SE, Bq, 0,  1.0f, 1.0f);
      // ifft -> real part into stacked slot l=0 (row stride 3E)
      k_fft2048<<<dim3(Bq * Eq), dim3(256), 0, stream>>>(atre, atim, stacked, nullptr,
          1, 1.0f / 2048.0f, Eq, SE, (long long)3 * Eq, (long long)Sq * 3 * Eq);
    } else if (d == 1) {  // ---- Wavelet ----
      k_haar2048<<<dim3(Bq * Eq), dim3(256), 0, stream>>>(xd, sre);
      k_conv3<<<dim3(Bq * Sq * Cq / 8), dim3(256), 0, stream>>>(sre, Wc[d], bc[d], chre);
      k_mix<<<dim3((unsigned)(MSE / 256)), dim3(256), 0, stream>>>(chre, fwb, c2re);
      gemm(c2re, c2re, scre, nullptr, Sq, Sq, Eq, Eq, Eq, Sq, SE, SE, SS, Bq, 1, iscale, 0.0f);
      k_row_softmax<<<dim3(Bq * Sq), dim3(256), 0, stream>>>(scre, wre, Sq);
      // w @ v -> stacked slot l=1
      gemm(wre, c2re, stacked + Eq, nullptr, Sq, Eq, Sq, Sq, Eq, 3 * Eq,
           SS, SE, (long long)Sq * 3 * Eq, Bq, 0, 1.0f, 0.0f);
    } else {  // ---- Cosine ----
      k_dct_mats<<<dim3((unsigned)(SS / 256)), dim3(256), 0, stream>>>(Dbf, Ibf);
      cvt(xd, xdbf, MSE);
      gemm(Dbf, xdbf, sre, nullptr, Sq, Eq, Sq, Sq, Eq, Eq, 0, SE, SE, Bq, 0, 1.0f, 0.0f);
      k_conv3<<<dim3(Bq * Sq * Cq / 8), dim3(256), 0, stream>>>(sre, Wc[d], bc[d], chre);
      k_mix<<<dim3((unsigned)(MSE / 256)), dim3(256), 0, stream>>>(chre, fwb, c2re);
      gemm(c2re, c2re, scre, nullptr, Sq, Sq, Eq, Eq, Eq, Sq, SE, SE, SS, Bq, 1, iscale, 0.0f);
      k_row_softmax<<<dim3(Bq * Sq), dim3(256), 0, stream>>>(scre, wre, Sq);
      gemm(wre, c2re, atre, nullptr, Sq, Eq, Sq, Sq, Eq, Eq, SS, SE, SE, Bq, 0, 1.0f, 0.0f);
      cvt(atre, atbf, MSE);
      // I @ out -> stacked slot l=2
      gemm(Ibf, atbf, stacked + 2 * Eq, nullptr, Sq, Eq, Sq, Sq, Eq, 3 * Eq,
           0, SE, (long long)Sq * 3 * Eq, Bq, 0, 1.0f, 0.0f);
    }
  }

  // ---- MHA over the 3 stacked tokens + fusion ----
  cvt(stacked, stackedbf, MSE * 3);
  cvt(mha_iw, wtbf, (long long)3 * Eq * Eq);
  gemm(stackedbf, wtbf, qkvb, mha_ib, Bq * Sq * 3, 3 * Eq, Eq, Eq, Eq, 3 * Eq,
       0, 0, 0, 1, 1, 1.0f, 0.0f);
  k_mha3<<<dim3(Bq * Sq * Hq / 256), dim3(256), 0, stream>>>(qkvb, obf);
  cvt(mha_ow, wtbf, (long long)Eq * Eq);
  gemm(obf, wtbf, corr, mha_ob, Bq * Sq * 3, Eq, Eq, Eq, Eq, Eq, 0, 0, 0, 1, 1, 1.0f, 0.0f);
  // corr (12288x1024) is exactly fused (4096x3072) in memory
  cvt(corr, corrbf, MSE * 3);
  cvt(fus_W, wtbf, (long long)3 * Eq * Eq);
  gemm(corrbf, wtbf, out, fus_b, Bq * Sq, Eq, 3 * Eq, 3 * Eq, 3 * Eq, Eq,
       0, 0, 0, 1, 1, 1.0f, 0.0f);
}
